// EdgeDet_47871705481620
// MI455X (gfx1250) — matrix operand order repaired and checked by reading, not compiled, not run on hardware
//
#include <hip/hip_runtime.h>
#include <hip/hip_bf16.h>
#include <math.h>

// ---------------------------------------------------------------------------
// CDNA5 (gfx1250) DGCNN forward. wave32.
//  - kNN Gram matrices: exact f32 WMMA (v_wmma_f32_16x16x4_f32), 2 j-tiles in
//    flight per iteration for memory-level parallelism.
//  - conv / phi / fc GEMMs: v_wmma_f32_16x16x32_f16 (f16 in, f32 accum).
//  - Tensor Data Mover: gather-mode tensor_load_to_lds pulls the 16 neighbor
//    rows straight into LDS (TENSORcnt); 1-D TDM tile load for the kNN i-tile.
// ---------------------------------------------------------------------------

typedef __attribute__((ext_vector_type(16))) _Float16 v16h;
typedef __attribute__((ext_vector_type(8)))  _Float16 v8h;
typedef __attribute__((ext_vector_type(2)))  _Float16 h2;
typedef __attribute__((ext_vector_type(8)))  float    v8f;
typedef __attribute__((ext_vector_type(2)))  float    v2f;
typedef __attribute__((ext_vector_type(4)))  unsigned int u32x4;
typedef __attribute__((ext_vector_type(4)))  int      i32x4;
typedef __attribute__((ext_vector_type(8)))  int      i32x8;

#define NB        4
#define KNB       16
#define MAXC      512
#define BN_EPS    1e-5f
#define NEG_SLOPE 0.2f

#if defined(__gfx1250__) && __has_builtin(__builtin_amdgcn_tensor_load_to_lds) && \
    __has_builtin(__builtin_amdgcn_s_wait_tensorcnt)
#define USE_TDM 1
#else
#define USE_TDM 0
#endif

// ---- WMMA fragment helpers -------------------------------------------------
__device__ __forceinline__ v16h load_a_frag(const _Float16* rowp, int hi) {
  v8h lo = *(const v8h*)(rowp + 8 * hi);
  v8h h8 = *(const v8h*)(rowp + 16 + 8 * hi);
  v16h a;
#pragma unroll
  for (int e = 0; e < 8; ++e) { a[e] = lo[e]; a[e + 8] = h8[e]; }
  return a;
}

__device__ __forceinline__ v8f zero8() {
  v8f c;
#pragma unroll
  for (int r = 0; r < 8; ++r) c[r] = 0.0f;
  return c;
}

#if USE_TDM
// ---- Tensor Data Mover descriptor builders (CDNA5 ISA §8) ------------------
// group1 bitfields: [17:16]=data_size  [79:48]=tensor_dim0  [111:80]=tensor_dim1
// [127:112]=tile_dim0  [143:128]=tile_dim1(#indices in gather)  [159:144]=tile_dim2
// [207:160]=tensor_dim0_stride  [255:208]=tensor_dim1_stride
__device__ __forceinline__ void tdm_build_g1(unsigned* w, unsigned ds, unsigned tdim0,
                                             unsigned tdim1, unsigned tile0, unsigned tile1,
                                             unsigned tile2, unsigned long long s0,
                                             unsigned long long s1) {
  w[0] = ds << 16;
  w[1] = (tdim0 & 0xFFFFu) << 16;
  w[2] = (tdim0 >> 16) | ((tdim1 & 0xFFFFu) << 16);
  w[3] = (tdim1 >> 16) | (tile0 << 16);
  w[4] = (tile1 & 0xFFFFu) | (tile2 << 16);
  w[5] = (unsigned)s0;
  w[6] = (unsigned)((s0 >> 32) & 0xFFFFu) | (((unsigned)s1 & 0xFFFFu) << 16);
  w[7] = (unsigned)(s1 >> 16);
}

// group0: [1:0]=count=1  [31]=gather_mode  [63:32]=lds_addr  [120:64]=global_addr
// [127:126]=type=2
__device__ __forceinline__ void tdm_issue(unsigned flags, unsigned lds_addr,
                                          unsigned long long ga, const unsigned* g1w,
                                          const unsigned* g2w, const unsigned* g3w) {
  u32x4 g0;
  g0[0] = 1u | flags;
  g0[1] = lds_addr;
  g0[2] = (unsigned)ga;
  g0[3] = (unsigned)((ga >> 32) & 0x01FFFFFFu) | (2u << 30);
  i32x8 g1;
#pragma unroll
  for (int e = 0; e < 8; ++e) g1[e] = (int)g1w[e];
  i32x4 g2, g3;
#pragma unroll
  for (int e = 0; e < 4; ++e) { g2[e] = (int)g2w[e]; g3[e] = (int)g3w[e]; }
  i32x8 g4z;
#pragma unroll
  for (int e = 0; e < 8; ++e) g4z[e] = 0;
  // clang-23 / amdgpu-toolchain 6-arg form: (g0, g1, g2, g3, <unused 256b>, cpol)
  __builtin_amdgcn_tensor_load_to_lds(g0, g1, g2, g3, g4z, 0);
}
#endif  // USE_TDM

// ---- small elementwise kernels --------------------------------------------
__global__ void padx_kernel(const float* __restrict__ x, float* __restrict__ out, int rows) {
  int t = blockIdx.x * blockDim.x + threadIdx.x;
  if (t >= rows * 4) return;
  int row = t >> 2, c = t & 3;
  out[t] = (c < 3) ? x[row * 3 + c] : 0.0f;
}

__global__ void packw_kernel(const float* __restrict__ w, _Float16* __restrict__ out,
                             int C, int D, int Cp) {  // w:[C,D] -> out:[D,Cp] f16
  int t = blockIdx.x * blockDim.x + threadIdx.x;
  if (t >= D * Cp) return;
  int d = t / Cp, c = t % Cp;
  out[t] = (c < C) ? (_Float16)w[(size_t)c * D + d] : (_Float16)0.0f;
}

__global__ void norms_kernel(const float* __restrict__ h, float* __restrict__ norms,
                             int rows, int Cs) {
  int t = blockIdx.x * blockDim.x + threadIdx.x;
  if (t >= rows) return;
  const float* r = h + (size_t)t * Cs;
  float s = 0.0f;
  for (int c = 0; c < Cs; ++c) s += r[c] * r[c];
  norms[t] = s;
}

__global__ void cvt_f16_kernel(const float* __restrict__ h, _Float16* __restrict__ out,
                               int rows, int Cs, int C, int Cp) {
  int t = blockIdx.x * blockDim.x + threadIdx.x;
  if (t >= rows * Cp) return;
  int row = t / Cp, c = t % Cp;
  out[t] = (c < C) ? (_Float16)h[(size_t)row * Cs + c] : (_Float16)0.0f;
}

__global__ void gather_kernel(const float* __restrict__ in, const int* __restrict__ ind,
                              float* __restrict__ out, int Nin, int Nout, int C) {
  int t = blockIdx.x * blockDim.x + threadIdx.x;
  if (t >= NB * Nout * C) return;
  int c = t % C, m = (t / C) % Nout, b = t / (C * Nout);
  int src = ind[b * Nout + m];
  out[t] = in[((size_t)b * Nin + src) * C + c];
}

// ---- kNN: exact f32 Gram via v_wmma_f32_16x16x4_f32 + top-16 --------------
// one wave per 16-row i-tile; two j-tiles in flight per iteration.
__global__ void knn_kernel(const float* __restrict__ h, const float* __restrict__ norms,
                           int N, int Cs, int* __restrict__ idxout) {
  __shared__ __align__(16) float sA[16 * MAXC];
  __shared__ float sD[16 * 32];
  __shared__ float sNi[16];
  int ntiles = N >> 4;
  int b = blockIdx.x / ntiles, it = blockIdx.x % ntiles;
  int i0 = it * 16;
  int l = threadIdx.x, n = l & 15, hi = l >> 4;
  const float* hb = h + (size_t)b * N * Cs;

#if USE_TDM
  {  // 1-D TDM tile: 16 contiguous rows of Cs f32 -> sA
    unsigned g1w[8];
    unsigned e16 = (unsigned)(16 * Cs);
    tdm_build_g1(g1w, /*ds=4B*/2, e16, 1u, e16, 0u, 0u, (unsigned long long)e16, 0ull);
    unsigned z4[4] = {0u, 0u, 0u, 0u};
    tdm_issue(0u, (unsigned)(unsigned long long)(uintptr_t)&sA[0],
              (unsigned long long)(uintptr_t)(hb + (size_t)i0 * Cs), g1w, z4, z4);
    __builtin_amdgcn_s_wait_tensorcnt(0);
    asm volatile("" ::: "memory");
  }
#else
  for (int r = 0; r < 16; ++r)
    for (int c = l; c < Cs; c += 32)
      sA[r * Cs + c] = hb[(size_t)(i0 + r) * Cs + c];
#endif
  if (l < 16) sNi[l] = norms[(size_t)b * N + i0 + l];
  __syncthreads();

  float bd[16]; int bj[16];
#pragma unroll
  for (int r = 0; r < 16; ++r) { bd[r] = 3.4e38f; bj[r] = 0; }

  int nk = Cs >> 2;
  for (int jt = 0; jt < ntiles; jt += 2) {
    int j0 = jt * 16, j1 = j0 + 16;
    float nj0v = norms[(size_t)b * N + j0 + n];
    float nj1v = norms[(size_t)b * N + j1 + n];
    v8f c0 = zero8(), c1 = zero8();
#if __has_builtin(__builtin_amdgcn_wmma_f32_16x16x4_f32)
    for (int kt = 0; kt < nk; ++kt) {
      int k0 = kt * 4 + 2 * hi;
      v2f a  = *(const v2f*)(&sA[n * Cs + k0]);
      v2f b0 = *(const v2f*)(&hb[(size_t)(j0 + n) * Cs + k0]);
      v2f b1 = *(const v2f*)(&hb[(size_t)(j1 + n) * Cs + k0]);
      c0 = __builtin_amdgcn_wmma_f32_16x16x4_f32(false, a, false, b0, (short)0, c0, false, false);
      c1 = __builtin_amdgcn_wmma_f32_16x16x4_f32(false, a, false, b1, (short)0, c1, false, false);
    }
#else
#pragma unroll
    for (int r = 0; r < 8; ++r) {
      int m = r + 8 * hi; float s0 = 0.0f, s1 = 0.0f;
      for (int cc = 0; cc < Cs; ++cc) {
        s0 += sA[m * Cs + cc] * hb[(size_t)(j0 + n) * Cs + cc];
        s1 += sA[m * Cs + cc] * hb[(size_t)(j1 + n) * Cs + cc];
      }
      c0[r] = s0; c1[r] = s1;
    }
#endif
    __syncthreads();  // previous pair's consumers done
#pragma unroll
    for (int r = 0; r < 8; ++r) {
      int m = r + 8 * hi;  // C-frag: lane holds col n, rows r+8*hi
      sD[m * 32 + n]      = sNi[m] + nj0v - 2.0f * c0[r];
      sD[m * 32 + 16 + n] = sNi[m] + nj1v - 2.0f * c1[r];
    }
    __syncthreads();
    if (l < 16) {  // lane l owns row i0+l
#pragma unroll 1
      for (int t = 0; t < 32; ++t) {
        float dv = sD[l * 32 + t];
        int jj = j0 + t;
        if (dv < bd[15]) {
          bd[15] = dv; bj[15] = jj;
#pragma unroll
          for (int r = 15; r > 0; --r)
            if (bd[r] < bd[r - 1]) {
              float td = bd[r]; bd[r] = bd[r - 1]; bd[r - 1] = td;
              int   tj = bj[r]; bj[r] = bj[r - 1]; bj[r - 1] = tj;
            }
        }
      }
    }
  }
  if (l < 16) {
    int* op = idxout + (size_t)(b * N + i0 + l) * KNB;
#pragma unroll
    for (int r = 0; r < 16; ++r) op[r] = bj[r];
  }
}

// ---- dense GEMM (rows x Cp) * (Cp x D) via v_wmma_f32_16x16x32_f16 --------
__global__ void gemm_rows16(const _Float16* __restrict__ A, const _Float16* __restrict__ WT,
                            const float* __restrict__ bias1, const float* __restrict__ bias2,
                            float* __restrict__ out, int rows, int Cp, int D, int mode) {
  int mt = blockIdx.x, dt = blockIdx.y;
  int l = threadIdx.x, n = l & 15, hi = l >> 4;
  const _Float16* arow = A + (size_t)(mt * 16 + n) * Cp;
  const _Float16* brow = WT + (size_t)(dt * 16 + n) * Cp;
  v8f c = zero8();
  int nk = Cp >> 5;
  for (int kt = 0; kt < nk; ++kt) {
    v16h a = load_a_frag(arow + kt * 32, hi);
    v16h bf = load_a_frag(brow + kt * 32, hi);
    c = __builtin_amdgcn_wmma_f32_16x16x32_f16(false, a, false, bf, (short)0, c, false, false);
  }
  int d = dt * 16 + n;
  float bb = bias1[d] + (bias2 ? bias2[d] : 0.0f);
#pragma unroll
  for (int r = 0; r < 8; ++r) {
    int m = r + 8 * hi;
    float v = c[r] + bb;
    if (mode == 1) v = 1.0f / (1.0f + __expf(-v));
    out[(size_t)(mt * 16 + m) * D + d] = v;
  }
}

// ---- EdgeConv: one wave per point; TDM gather of 16 neighbor rows ----------
__global__ void edgeconv_kernel(const _Float16* __restrict__ hf, const int* __restrict__ nbr,
                                const float* __restrict__ P, const _Float16* __restrict__ WT,
                                const float* __restrict__ g, const float* __restrict__ bb,
                                float* __restrict__ hout, int N, int Cp, int D) {
  __shared__ __align__(16) _Float16 sE[16 * MAXC];
  int p = blockIdx.x;  // flattened b*N + i
  int l = threadIdx.x, n = l & 15, hi = l >> 4;
  const _Float16* hrow = hf + (size_t)p * Cp;
  const _Float16* hbat = hf + (size_t)(p / N) * N * Cp;

#if USE_TDM
  {  // gather-mode TDM: 16 rows of Cp f16 selected by 16-bit neighbor indices
    const int* ip = nbr + (size_t)p * KNB;
    unsigned g2w[4], g3w[4];
#pragma unroll
    for (int k2 = 0; k2 < 4; ++k2) {
      g2w[k2] = ((unsigned)ip[2 * k2] & 0xFFFFu) | (((unsigned)ip[2 * k2 + 1] & 0xFFFFu) << 16);
      g3w[k2] = ((unsigned)ip[8 + 2 * k2] & 0xFFFFu) | (((unsigned)ip[9 + 2 * k2] & 0xFFFFu) << 16);
    }
    unsigned g1w[8];
    tdm_build_g1(g1w, /*ds=2B*/1, (unsigned)Cp, (unsigned)N, (unsigned)Cp, 16u, 0u,
                 (unsigned long long)Cp, 0ull);
    tdm_issue(0x80000000u /*gather, 16-bit idx*/,
              (unsigned)(unsigned long long)(uintptr_t)&sE[0],
              (unsigned long long)(uintptr_t)hbat, g1w, g2w, g3w);
    __builtin_amdgcn_s_wait_tensorcnt(0);
    asm volatile("" ::: "memory");
    // E = nbr - center, in place (packed f16)
    h2* sE2 = (h2*)sE;
    int half = Cp >> 1;
    for (int q = l; q < 16 * half; q += 32) {
      int tt = q - (q / half) * half;
      h2 ci = *(const h2*)(hrow + 2 * tt);
      sE2[q] = sE2[q] - ci;
    }
    __syncthreads();
  }
#else
  {
    __shared__ int sN[16];
    if (l < 16) sN[l] = nbr[(size_t)p * KNB + l];
    __syncthreads();
    for (int j = 0; j < 16; ++j) {
      const _Float16* hj = hbat + (size_t)sN[j] * Cp;
      for (int t = l; t < (Cp >> 1); t += 32) {
        h2 a = *(const h2*)(hj + 2 * t);
        h2 ci = *(const h2*)(hrow + 2 * t);
        h2 e = a - ci;
        *(h2*)(&sE[j * Cp + 2 * t]) = e;
      }
    }
    __syncthreads();
  }
#endif

  const float rs = rsqrtf(1.0f + BN_EPS);
  const float* Prow = P + (size_t)p * D;
  int nk = Cp >> 5, nd = D >> 4;
  for (int dt = 0; dt < nd; ++dt) {
    const _Float16* brow = WT + (size_t)(dt * 16 + n) * Cp;
    v8f c = zero8();
    for (int kt = 0; kt < nk; ++kt) {
      v16h a = load_a_frag(&sE[n * Cp + kt * 32], hi);  // A rows = neighbors
      v16h bf = load_a_frag(brow + kt * 32, hi);
      c = __builtin_amdgcn_wmma_f32_16x16x32_f16(false, a, false, bf, (short)0, c, false, false);
    }
    int d = dt * 16 + n;
    float scale = g[d] * rs, shift = bb[d], base = Prow[d];
    float vmax = -3.4e38f;
#pragma unroll
    for (int r = 0; r < 8; ++r) {
      float v = (c[r] + base) * scale + shift;
      vmax = fmaxf(vmax, v);
    }
    float o = __shfl_xor(vmax, 16, 32);
    vmax = fmaxf(vmax, o);
    vmax = vmax > 0.0f ? vmax : NEG_SLOPE * vmax;
    if (hi == 0) hout[(size_t)p * D + d] = vmax;
  }
}

// ---------------------------------------------------------------------------
// Host orchestration
// ---------------------------------------------------------------------------
static const int   Ns_[8]  = {4096, 4096, 1024, 1024, 1024, 256, 256, 256};
static const int   Cin_[8] = {3, 32, 64, 128, 128, 256, 512, 512};
static const int   Cs_[8]  = {4, 32, 64, 128, 128, 256, 512, 512};
static const int   Cp_[8]  = {32, 32, 64, 128, 128, 256, 512, 512};
static const int   Do_[8]  = {32, 64, 128, 128, 256, 512, 512, 1024};
static const size_t wOff_[8] = {0, 1024, 3072, 11264, 27648, 60416, 191488, 453632};

extern "C" void kernel_launch(void* const* d_in, const int* in_sizes, int n_in,
                              void* d_out, int out_size, void* d_ws, size_t ws_size,
                              hipStream_t stream) {
  (void)in_sizes; (void)n_in; (void)out_size; (void)ws_size;
  const float* x = (const float*)d_in[0];
  const int* ind[3] = {(const int*)d_in[1], (const int*)d_in[2], (const int*)d_in[3]};
  const float *tw[8], *tb[8], *pw[8], *pb[8], *bg[8], *bbv[8];
  for (int i = 0; i < 8; ++i) {
    tw[i]  = (const float*)d_in[4 + i];
    tb[i]  = (const float*)d_in[12 + i];
    pw[i]  = (const float*)d_in[20 + i];
    pb[i]  = (const float*)d_in[28 + i];
    bg[i]  = (const float*)d_in[36 + i];
    bbv[i] = (const float*)d_in[44 + i];
  }
  const float* fcw = (const float*)d_in[52];
  const float* fcb = (const float*)d_in[53];

  const size_t MB = (size_t)1 << 20;
  char* ws = (char*)d_ws;
  float*    hA   = (float*)(ws + 0);             // 4 MB
  float*    hB   = (float*)(ws + 4 * MB);        // 4 MB
  float*    Pbuf = (float*)(ws + 8 * MB);        // 4 MB
  _Float16* hf16 = (_Float16*)(ws + 12 * MB);    // 2 MB
  int*      idxb = (int*)(ws + 14 * MB);         // 1 MB
  float*    nrm  = (float*)(ws + 15 * MB);       // 256 KB
  _Float16* wth  = (_Float16*)(ws + 15 * MB + 256 * 1024);  // 2 MB
  _Float16* wph  = (_Float16*)((char*)wth + 2 * MB);        // 2 MB
  _Float16* wfc  = (_Float16*)((char*)wph + 2 * MB);        // 128 KB

  const int TB = 256;
  padx_kernel<<<(NB * 4096 * 4 + TB - 1) / TB, TB, 0, stream>>>(x, hA, NB * 4096);
  for (int li = 0; li < 8; ++li) {
    int C = Cin_[li], D = Do_[li], Cp = Cp_[li];
    packw_kernel<<<(D * Cp + TB - 1) / TB, TB, 0, stream>>>(tw[li], wth + wOff_[li], C, D, Cp);
    packw_kernel<<<(D * Cp + TB - 1) / TB, TB, 0, stream>>>(pw[li], wph + wOff_[li], C, D, Cp);
  }
  packw_kernel<<<(64 * 1024 + TB - 1) / TB, TB, 0, stream>>>(fcw, wfc, 1024, 64, 1024);

  float* hcur = hA;
  float* hnext = hB;
  for (int li = 0; li < 8; ++li) {
    int N = Ns_[li], C = Cin_[li], Cs = Cs_[li], Cp = Cp_[li], D = Do_[li];
    int rows = NB * N;
    norms_kernel<<<(rows + TB - 1) / TB, TB, 0, stream>>>(hcur, nrm, rows, Cs);
    knn_kernel<<<NB * (N / 16), 32, 0, stream>>>(hcur, nrm, N, Cs, idxb);
    cvt_f16_kernel<<<(rows * Cp + TB - 1) / TB, TB, 0, stream>>>(hcur, hf16, rows, Cs, C, Cp);
    gemm_rows16<<<dim3(rows / 16, D / 16), 32, 0, stream>>>(
        hf16, wph + wOff_[li], pb[li], tb[li], Pbuf, rows, Cp, D, 0);
    edgeconv_kernel<<<rows, 32, 0, stream>>>(
        hf16, idxb, Pbuf, wth + wOff_[li], bg[li], bbv[li], hnext, N, Cp, D);
    { float* t = hcur; hcur = hnext; hnext = t; }
    if (li == 1) {
      gather_kernel<<<(NB * 1024 * 64 + TB - 1) / TB, TB, 0, stream>>>(hcur, ind[0], hnext, 4096, 1024, 64);
      float* t = hcur; hcur = hnext; hnext = t;
    } else if (li == 4) {
      gather_kernel<<<(NB * 256 * 256 + TB - 1) / TB, TB, 0, stream>>>(hcur, ind[1], hnext, 1024, 256, 256);
      float* t = hcur; hcur = hnext; hnext = t;
    } else if (li == 7) {
      gather_kernel<<<(NB * 128 * 1024 + TB - 1) / TB, TB, 0, stream>>>(hcur, ind[2], hnext, 256, 128, 1024);
      float* t = hcur; hcur = hnext; hnext = t;
    }
  }
  cvt_f16_kernel<<<(512 * 1024 + TB - 1) / TB, TB, 0, stream>>>(hcur, hf16, 512, 1024, 1024, 1024);
  gemm_rows16<<<dim3(512 / 16, 64 / 16), 32, 0, stream>>>(
      hf16, wfc, fcb, nullptr, (float*)d_out, 512, 1024, 64, 1);
}
